// MIOSTONEModel_82197084111621
// MI455X (gfx1250) — matrix operand order repaired and checked by reading, not compiled, not run on hardware
//
#include <hip/hip_runtime.h>
#include <hip/hip_bf16.h>

typedef __attribute__((ext_vector_type(16))) _Float16 v16h;
typedef __attribute__((ext_vector_type(8)))  float    v8f;
typedef __attribute__((ext_vector_type(4)))  float    f4;

union Frag16 { v16h v; _Float16 h[16]; };

// A fragment: 16x32 (MxK) f16 tile per ISA layout.
// lane<16: M=l15, K in {k0..k0+7, k0+16..k0+23}; lane>=16: K shifted by 8.
template <int KR>
__device__ __forceinline__ v16h load_a_frag(const float* __restrict__ base, int lda,
                                            int mt, int l15, int hi, int kc) {
  Frag16 f;
  const float* rp = base + (size_t)(mt * 16 + l15) * (size_t)lda;
  const int k0 = kc * 32 + hi * 8;
#pragma unroll
  for (int g = 0; g < 2; ++g) {
    const int kb = k0 + g * 16;
    if ((KR % 32 == 0) || kb < KR) {   // folds: KR=256 -> always; KR=8 -> hi==0 && g==0 && kc==0
      f4 a = *(const f4*)(rp + kb);
      f4 b = *(const f4*)(rp + kb + 4);
      f.h[g*8+0] = (_Float16)a.x; f.h[g*8+1] = (_Float16)a.y;
      f.h[g*8+2] = (_Float16)a.z; f.h[g*8+3] = (_Float16)a.w;
      f.h[g*8+4] = (_Float16)b.x; f.h[g*8+5] = (_Float16)b.y;
      f.h[g*8+6] = (_Float16)b.z; f.h[g*8+7] = (_Float16)b.w;
    } else {
#pragma unroll
      for (int i = 0; i < 8; ++i) f.h[g*8+i] = (_Float16)0.0f;
    }
  }
  return f.v;
}

// B fragment: 32x16 (KxN) f16 tile; B[k][n] = W[rowbase+n][colbase+k].
// lane<16: N=l15, K in {k0..k0+7, k0+16..k0+23}; lane>=16: K shifted by 8.
template <int KR>
__device__ __forceinline__ v16h load_b_frag(const float* __restrict__ W, int ldw,
                                            int rowbase, size_t colbase,
                                            int nt, int l15, int hi, int kc) {
  Frag16 f;
  const float* rp = W + (size_t)(rowbase + nt * 16 + l15) * (size_t)ldw + colbase;
  const int k0 = kc * 32 + hi * 8;
#pragma unroll
  for (int g = 0; g < 2; ++g) {
    const int kb = k0 + g * 16;
    if ((KR % 32 == 0) || kb < KR) {
      f4 a = *(const f4*)(rp + kb);
      f4 b = *(const f4*)(rp + kb + 4);
      f.h[g*8+0] = (_Float16)a.x; f.h[g*8+1] = (_Float16)a.y;
      f.h[g*8+2] = (_Float16)a.z; f.h[g*8+3] = (_Float16)a.w;
      f.h[g*8+4] = (_Float16)b.x; f.h[g*8+5] = (_Float16)b.y;
      f.h[g*8+6] = (_Float16)b.z; f.h[g*8+7] = (_Float16)b.w;
    } else {
#pragma unroll
      for (int i = 0; i < 8; ++i) f.h[g*8+i] = (_Float16)0.0f;
    }
  }
  return f.v;
}

#define WMMA_F16(A, B, C) \
  __builtin_amdgcn_wmma_f32_16x16x32_f16(false, (A), false, (B), (short)0, (C), false, false)

// One tree node per block. 4 waves: wave = {branch(mlp/linear)} x {nt(N-half)}.
// Each wave computes both M-tiles of a 32x16 slab of one branch's 32x32xKR GEMM,
// dumps raw f32 accumulators to LDS, then all 128 threads do bias/ReLU/gate.
template <int KR>
__global__ __launch_bounds__(128, 1)
void tree_layer_kernel(const float* __restrict__ Ax,
                       const float* __restrict__ Al, int lda,
                       const float* __restrict__ Wm,
                       const float* __restrict__ bm,
                       const float* __restrict__ Wl,
                       const float* __restrict__ bl, int ldw,
                       const float* __restrict__ gate,
                       float* __restrict__ xout,
                       float* __restrict__ xlout, int ldo) {
  __shared__ float Acc[2][32][32];            // [branch][m][n], raw accumulators

  const int j      = blockIdx.x;              // node index
  const int lane   = threadIdx.x & 31;
  const int wave   = threadIdx.x >> 5;        // 0..3
  const int branch = wave >> 1;               // 0 = mlp, 1 = linear
  const int nt     = wave & 1;                // N-half
  const int l15    = lane & 15, hi = lane >> 4;

  const float* Abase = (branch ? Al : Ax) + (size_t)KR * j;
  const float* Wb    = branch ? Wl : Wm;
  const size_t colbase = (size_t)KR * j;
  const int rowbase = 32 * j;

  v8f acc0 = (v8f)0.0f, acc1 = (v8f)0.0f;

#pragma unroll 4
  for (int kc = 0; kc < (KR + 31) / 32; ++kc) {
    v16h a0 = load_a_frag<KR>(Abase, lda, 0, l15, hi, kc);
    v16h a1 = load_a_frag<KR>(Abase, lda, 1, l15, hi, kc);
    v16h b  = load_b_frag<KR>(Wb, ldw, rowbase, colbase, nt, l15, hi, kc);
    acc0 = WMMA_F16(a0, b, acc0);
    acc1 = WMMA_F16(a1, b, acc1);
  }

  // C/D layout: VGPR r holds M = mt*16 + hi*8 + r, N = nt*16 + l15.
  const int n = nt * 16 + l15;
#pragma unroll
  for (int r = 0; r < 8; ++r) {
    Acc[branch][hi * 8 + r][n]      = acc0[r];
    Acc[branch][16 + hi * 8 + r][n] = acc1[r];
  }
  __syncthreads();

  const float g = gate[0], gi = 1.0f - g;
#pragma unroll
  for (int i = 0; i < 8; ++i) {
    const int idx = i * 128 + threadIdx.x;    // 1024 elements, n-contiguous
    const int m = idx >> 5, nn = idx & 31;
    float hv  = Acc[0][m][nn] + bm[rowbase + nn];
    hv = hv > 0.0f ? hv : 0.0f;
    float xlv = Acc[1][m][nn] + bl[rowbase + nn];
    const size_t o = (size_t)m * (size_t)ldo + (size_t)rowbase + nn;
    xlout[o] = xlv;
    xout[o]  = g * hv + gi * xlv;
  }
}

// Root node (KR=256) + BatchNorm (training-mode, biased var) + out projection.
// Single wave; branches computed sequentially to keep VGPR pressure low.
__global__ __launch_bounds__(32, 1)
void tree_final_kernel(const float* __restrict__ x1,
                       const float* __restrict__ xl1,
                       const float* __restrict__ Wm0,
                       const float* __restrict__ bm0,
                       const float* __restrict__ Wl0,
                       const float* __restrict__ bl0,
                       const float* __restrict__ gate,
                       const float* __restrict__ gamma,
                       const float* __restrict__ beta,
                       const float* __restrict__ Wout,
                       const float* __restrict__ bout,
                       float* __restrict__ out) {
  constexpr int KR = 256;
  __shared__ float Hm[32 * 32];
  __shared__ float X[32 * 32];
  const int lane = threadIdx.x & 31;
  const int l15 = lane & 15, hi = lane >> 4;

  // Pass 1: mlp branch accumulators -> LDS.
  {
    v8f acc[4];
#pragma unroll
    for (int t = 0; t < 4; ++t) acc[t] = (v8f)0.0f;
#pragma unroll 4
    for (int kc = 0; kc < 8; ++kc) {
      v16h a0 = load_a_frag<KR>(x1, KR, 0, l15, hi, kc);
      v16h a1 = load_a_frag<KR>(x1, KR, 1, l15, hi, kc);
#pragma unroll
      for (int ntile = 0; ntile < 2; ++ntile) {
        v16h b = load_b_frag<KR>(Wm0, KR, 0, 0, ntile, l15, hi, kc);
        acc[0 * 2 + ntile] = WMMA_F16(a0, b, acc[0 * 2 + ntile]);
        acc[1 * 2 + ntile] = WMMA_F16(a1, b, acc[1 * 2 + ntile]);
      }
    }
#pragma unroll
    for (int mt = 0; mt < 2; ++mt)
#pragma unroll
      for (int ntile = 0; ntile < 2; ++ntile) {
        const int n = ntile * 16 + l15;
#pragma unroll
        for (int r = 0; r < 8; ++r)
          Hm[(mt * 16 + hi * 8 + r) * 32 + n] = acc[mt * 2 + ntile][r];
      }
  }
  __syncthreads();

  // Pass 2: linear branch; combine with mlp result into X.
  {
    v8f acc[4];
#pragma unroll
    for (int t = 0; t < 4; ++t) acc[t] = (v8f)0.0f;
#pragma unroll 4
    for (int kc = 0; kc < 8; ++kc) {
      v16h a0 = load_a_frag<KR>(xl1, KR, 0, l15, hi, kc);
      v16h a1 = load_a_frag<KR>(xl1, KR, 1, l15, hi, kc);
#pragma unroll
      for (int ntile = 0; ntile < 2; ++ntile) {
        v16h b = load_b_frag<KR>(Wl0, KR, 0, 0, ntile, l15, hi, kc);
        acc[0 * 2 + ntile] = WMMA_F16(a0, b, acc[0 * 2 + ntile]);
        acc[1 * 2 + ntile] = WMMA_F16(a1, b, acc[1 * 2 + ntile]);
      }
    }
    const float g = gate[0], gi = 1.0f - g;
#pragma unroll
    for (int mt = 0; mt < 2; ++mt)
#pragma unroll
      for (int ntile = 0; ntile < 2; ++ntile) {
        const int n = ntile * 16 + l15;
        const float bmv = bm0[n], blv = bl0[n];
#pragma unroll
        for (int r = 0; r < 8; ++r) {
          const int m = mt * 16 + hi * 8 + r;
          float hv = Hm[m * 32 + n] + bmv;
          hv = hv > 0.0f ? hv : 0.0f;
          float xlv = acc[mt * 2 + ntile][r] + blv;
          X[m * 32 + n] = g * hv + gi * xlv;
        }
      }
  }
  __syncthreads();

  // BatchNorm: one lane per feature column.
  {
    const int n = lane;
    float mu = 0.0f;
#pragma unroll
    for (int m = 0; m < 32; ++m) mu += X[m * 32 + n];
    mu *= (1.0f / 32.0f);
    float var = 0.0f;
#pragma unroll
    for (int m = 0; m < 32; ++m) { float d = X[m * 32 + n] - mu; var += d * d; }
    var *= (1.0f / 32.0f);
    const float rs = rsqrtf(var + 1e-5f);
    const float ga = gamma[n], be = beta[n];
#pragma unroll
    for (int m = 0; m < 32; ++m)
      X[m * 32 + n] = (X[m * 32 + n] - mu) * rs * ga + be;
  }
  __syncthreads();

  // Output projection: one lane per batch row.
  {
    const int m = lane;
    float y0 = bout[0], y1 = bout[1];
#pragma unroll
    for (int n = 0; n < 32; ++n) {
      float v = X[m * 32 + n];
      y0 += v * Wout[n];
      y1 += v * Wout[32 + n];
    }
    out[m * 2 + 0] = y0;
    out[m * 2 + 1] = y1;
  }
}

extern "C" void kernel_launch(void* const* d_in, const int* in_sizes, int n_in,
                              void* d_out, int out_size, void* d_ws, size_t ws_size,
                              hipStream_t stream) {
  const float* x    = (const float*)d_in[0];
  const float* Wm3  = (const float*)d_in[1];
  const float* bm3  = (const float*)d_in[2];
  const float* Wl3  = (const float*)d_in[3];
  const float* bl3  = (const float*)d_in[4];
  const float* Wm2  = (const float*)d_in[5];
  const float* bm2  = (const float*)d_in[6];
  const float* Wl2  = (const float*)d_in[7];
  const float* bl2  = (const float*)d_in[8];
  const float* Wm1  = (const float*)d_in[9];
  const float* bm1  = (const float*)d_in[10];
  const float* Wl1  = (const float*)d_in[11];
  const float* bl1  = (const float*)d_in[12];
  const float* Wm0  = (const float*)d_in[13];
  const float* bm0  = (const float*)d_in[14];
  const float* Wl0  = (const float*)d_in[15];
  const float* bl0  = (const float*)d_in[16];
  const float* gate = (const float*)d_in[17];
  const float* gam  = (const float*)d_in[18];
  const float* bet  = (const float*)d_in[19];
  const float* Wout = (const float*)d_in[20];
  const float* bout = (const float*)d_in[21];
  float* out = (float*)d_out;

  // Workspace (floats): x3[32x16384] xl3[32x16384] x2[32x2048] xl2 x1[32x256] xl1
  float* ws  = (float*)d_ws;
  float* x3  = ws;
  float* xl3 = x3  + (size_t)32 * 16384;
  float* x2  = xl3 + (size_t)32 * 16384;
  float* xl2 = x2  + (size_t)32 * 2048;
  float* x1  = xl2 + (size_t)32 * 2048;
  float* xl1 = x1  + (size_t)32 * 256;

  // d=3: 512 nodes, KR=8; linear-branch input is x itself.
  tree_layer_kernel<8><<<512, 128, 0, stream>>>(x, x, 4096, Wm3, bm3, Wl3, bl3, 4096,
                                                gate, x3, xl3, 16384);
  // d=2: 64 nodes, KR=256 (dominant 4 MB weight read; 256 waves in flight).
  tree_layer_kernel<256><<<64, 128, 0, stream>>>(x3, xl3, 16384, Wm2, bm2, Wl2, bl2, 16384,
                                                 gate, x2, xl2, 2048);
  // d=1: 8 nodes, KR=256.
  tree_layer_kernel<256><<<8, 128, 0, stream>>>(x2, xl2, 2048, Wm1, bm1, Wl1, bl1, 2048,
                                                gate, x1, xl1, 256);
  // d=0 + BatchNorm + output projection.
  tree_final_kernel<<<1, 32, 0, stream>>>(x1, xl1, Wm0, bm0, Wl0, bl0,
                                          gate, gam, bet, Wout, bout, out);
}
